// MatryoshkaSampledSoftmaxLoss_23373212024870
// MI455X (gfx1250) — matrix operand order repaired and checked by reading, not compiled, not run on hardware
//
#include <hip/hip_runtime.h>
#include <math.h>

typedef __attribute__((ext_vector_type(16))) __bf16 v16bf;
typedef __attribute__((ext_vector_type(8)))  __bf16 v8bf;
typedef __attribute__((ext_vector_type(8)))  float  v8f;
typedef __attribute__((ext_vector_type(4)))  float  v4f;

#define NROWS  4096
#define DIM    1024
#define VOCAB  50257
#define VPAD   50272      // 3142 * 16
#define VTILES 3142
#define LOWR   64
#define NCAND  2048
#define CHUNK  32
#define NCHUNK 128
#define SCALE  20.0f
#define AUXW   0.2f

__device__ __forceinline__ float wave_sum(float v) {
    for (int off = 16; off > 0; off >>= 1) v += __shfl_xor(v, off, 32);
    return v;
}

// ---- embedding: row norms + normalized full row (bf16) + normalized low-rank row (bf16)
__global__ void emb_norm_kernel(const float* __restrict__ W,
                                float* __restrict__ inv_full,
                                float* __restrict__ inv_low,
                                __bf16* __restrict__ wlow,
                                __bf16* __restrict__ wnorm) {
    int wave = (blockIdx.x * blockDim.x + threadIdx.x) >> 5;
    int lane = threadIdx.x & 31;
    if (wave >= VPAD) return;
    int v = wave;
    if (v >= VOCAB) {                      // zero-pad rows so the scan GEMM reads zeros
        wlow[(size_t)v * LOWR + lane]      = (__bf16)0.0f;
        wlow[(size_t)v * LOWR + 32 + lane] = (__bf16)0.0f;
        return;
    }
    const float* row = W + (size_t)v * DIM;
    float x[32];
    float ssf = 0.f, ssl = 0.f;
    for (int i = 0; i < 32; ++i) {
        x[i] = row[i * 32 + lane];
        ssf += x[i] * x[i];
        if (i < 2) ssl += x[i] * x[i];
    }
    ssf = wave_sum(ssf);
    ssl = wave_sum(ssl);
    float invf = 1.0f / fmaxf(sqrtf(ssf), 1e-12f);
    float invl = 1.0f / fmaxf(sqrtf(ssl), 1e-12f);
    if (lane == 0) { inv_full[v] = invf; inv_low[v] = invl; }
    __bf16* on = wnorm + (size_t)v * DIM;
    for (int i = 0; i < 32; ++i) on[i * 32 + lane] = (__bf16)(x[i] * invf);
    wlow[(size_t)v * LOWR + lane]      = (__bf16)(x[0] * invl);
    wlow[(size_t)v * LOWR + 32 + lane] = (__bf16)(x[1] * invl);
}

// ---- hidden-state normalization (clip 1e-6): fp32 copy (for means / positives) + bf16 copy (WMMA A operand)
__global__ void h_norm_kernel(const float* __restrict__ H,
                              float* __restrict__ hf,
                              __bf16* __restrict__ hb) {
    int wave = (blockIdx.x * blockDim.x + threadIdx.x) >> 5;
    int lane = threadIdx.x & 31;
    if (wave >= NROWS) return;
    const float* row = H + (size_t)wave * DIM;
    float x[32];
    float ss = 0.f;
    for (int i = 0; i < 32; ++i) { x[i] = row[i * 32 + lane]; ss += x[i] * x[i]; }
    ss = wave_sum(ss);
    float inv = 1.0f / fmaxf(sqrtf(ss), 1e-6f);
    float*  of = hf + (size_t)wave * DIM;
    __bf16* ob = hb + (size_t)wave * DIM;
    for (int i = 0; i < 32; ++i) {
        float y = x[i] * inv;
        of[i * 32 + lane] = y;
        ob[i * 32 + lane] = (__bf16)y;
    }
}

// ---- per-chunk mean of low-rank slice (deterministic, no atomics)
__global__ void h_mean_kernel(const float* __restrict__ hf, float* __restrict__ hmean) {
    int c = blockIdx.x;          // 128 chunks
    int d = threadIdx.x;         // 64 dims
    float s = 0.f;
    for (int r = 0; r < CHUNK; ++r) s += hf[(size_t)(c * CHUNK + r) * DIM + d];
    hmean[c * LOWR + d] = s * (1.0f / (float)CHUNK);
}

// ---- scan GEMM: scan[c][v] = w_low_norm[v] . h_mean[c]  (K=64, bf16 WMMA)
__global__ void scan_kernel(const __bf16* __restrict__ wlow,
                            const float* __restrict__ hmean,
                            float* __restrict__ scan) {
    int lane = threadIdx.x & 31;
    int v0   = blockIdx.x * 16;
    int mrow = v0 + (lane & 15);
    int kbA  = (lane >> 4) * 16;
    int kbB  = (lane >> 4) * 16;

    v16bf A[2];
    for (int s = 0; s < 2; ++s) {
        v8bf lo = *(const v8bf*)(wlow + (size_t)mrow * LOWR + s * 32 + kbA);
        v8bf hi = *(const v8bf*)(wlow + (size_t)mrow * LOWR + s * 32 + kbA + 8);
        for (int i = 0; i < 8; ++i) { A[s][i] = lo[i]; A[s][8 + i] = hi[i]; }
    }
    for (int ct = 0; ct < 8; ++ct) {
        int n = ct * 16 + (lane & 15);         // chunk index
        v8f acc = {};
        for (int s = 0; s < 2; ++s) {
            const float* bp = hmean + n * LOWR + s * 32 + kbB;
            v16bf B;
            for (int i = 0; i < 16; ++i) B[i] = (__bf16)bp[i];
            acc = __builtin_amdgcn_wmma_f32_16x16x32_bf16(false, A[s], false, B,
                                                          (short)0, acc, false, false);
        }
        int mb = v0 + (lane >> 4) * 8;
        float* out = scan + (size_t)n * VPAD + mb;   // contiguous over 8 M values
        *(v4f*)(out)     = (v4f){acc[0], acc[1], acc[2], acc[3]};
        *(v4f*)(out + 4) = (v4f){acc[4], acc[5], acc[6], acc[7]};
    }
}

// ---- exact-threshold radix top-2048 (deterministic compaction)
__device__ __forceinline__ unsigned fkey(float x) {
    unsigned u = __float_as_uint(x);
    return (u & 0x80000000u) ? ~u : (u | 0x80000000u);   // monotonic order-preserving map
}

__global__ void topk_kernel(const float* __restrict__ scan, int* __restrict__ topidx) {
    const int c = blockIdx.x;
    const float* row = scan + (size_t)c * VPAD;
    __shared__ unsigned hist[4096];
    __shared__ unsigned thi[1024];
    __shared__ unsigned teq[1024];
    __shared__ unsigned sh_prefix, sh_krem;
    const int tid = threadIdx.x, nt = blockDim.x;

    if (tid == 0) { sh_prefix = 0u; sh_krem = NCAND; }
    __syncthreads();

    const int      shifts[3] = {20, 8, 0};
    const unsigned nbins[3]  = {4096u, 4096u, 256u};
    const unsigned pmasks[3] = {0u, 0xFFF00000u, 0xFFFFFF00u};

    for (int lev = 0; lev < 3; ++lev) {
        for (int b = tid; b < 4096; b += nt) hist[b] = 0u;
        __syncthreads();
        unsigned pref  = sh_prefix;
        unsigned pmask = pmasks[lev];
        int      sh    = shifts[lev];
        unsigned bmask = nbins[lev] - 1u;
        for (int j = tid; j < VOCAB; j += nt) {
            unsigned u = fkey(row[j]);
            if ((u & pmask) == pref) atomicAdd(&hist[(u >> sh) & bmask], 1u);
        }
        __syncthreads();
        if (tid == 0) {
            unsigned krem = sh_krem, cum = 0; int bsel = 0;
            for (int b = (int)nbins[lev] - 1; b >= 0; --b) {
                if (cum + hist[b] >= krem) { bsel = b; break; }
                cum += hist[b];
            }
            sh_krem   = krem - cum;
            sh_prefix = pref | ((unsigned)bsel << sh);
        }
        __syncthreads();
    }
    const unsigned T    = sh_prefix;          // exact threshold key
    const unsigned krem = sh_krem;            // # of ==T entries to take
    const unsigned nhi  = NCAND - krem;       // # of >T entries (exact)

    unsigned ch = 0, ce = 0;
    for (int j = tid; j < VOCAB; j += nt) {
        unsigned u = fkey(row[j]);
        ch += (u > T); ce += (u == T);
    }
    thi[tid] = ch; teq[tid] = ce;
    __syncthreads();
    if (tid == 0) {                           // serial exclusive scans -> deterministic positions
        unsigned run = 0;
        for (int i = 0; i < nt; ++i) { unsigned x = thi[i]; thi[i] = run; run += x; }
        run = 0;
        for (int i = 0; i < nt; ++i) { unsigned x = teq[i]; teq[i] = run; run += x; }
    }
    __syncthreads();
    int* out = topidx + c * NCAND;
    unsigned p = thi[tid], q = teq[tid];
    for (int j = tid; j < VOCAB; j += nt) {
        unsigned u = fkey(row[j]);
        if (u > T) { out[p++] = j; }
        else if (u == T) { if (q < krem) out[nhi + q] = j; q++; }
    }
}

// ---- fused gather+GEMM over prenormalized bf16 operands (full K=1024 and low-rank K=64)
__global__ void neg_gemm_kernel(const __bf16* __restrict__ hb,
                                const __bf16* __restrict__ wnorm,
                                const __bf16* __restrict__ wlow,
                                const int* __restrict__ topidx,
                                const int* __restrict__ tids,
                                float* __restrict__ negm,
                                float* __restrict__ nega) {
    int lane = threadIdx.x & 31;
    int bid  = blockIdx.x;
    int c  = bid >> 8;                 // 256 (=128 ntiles * 2 mtiles) tiles per chunk
    int r  = bid & 255;
    int ntile = r >> 1;
    int mtile = r & 1;

    int col = ntile * 16 + (lane & 15);
    int idx = topidx[c * NCAND + col];
    int arow = c * CHUNK + mtile * 16 + (lane & 15);
    int kbA = (lane >> 4) * 16;
    int kbB = (lane >> 4) * 16;
    const __bf16* Ab = hb    + (size_t)arow * DIM;
    const __bf16* Bb = wnorm + (size_t)idx  * DIM;

    v8f acc = {};
    for (int s = 0; s < 32; ++s) {
        int base = s * 32;
        v8bf alo = *(const v8bf*)(Ab + base + kbA);
        v8bf ahi = *(const v8bf*)(Ab + base + kbA + 8);
        v16bf A;
        for (int i = 0; i < 8; ++i) { A[i] = alo[i]; A[8 + i] = ahi[i]; }
        v16bf B = *(const v16bf*)(Bb + base + kbB);      // gathered, prenormalized
        acc = __builtin_amdgcn_wmma_f32_16x16x32_bf16(false, A, false, B,
                                                      (short)0, acc, false, false);
    }
    v8f accL = {};
    for (int s = 0; s < 2; ++s) {
        int base = s * 32;
        v8bf alo = *(const v8bf*)(Ab + base + kbA);
        v8bf ahi = *(const v8bf*)(Ab + base + kbA + 8);
        v16bf A;
        for (int i = 0; i < 8; ++i) { A[i] = alo[i]; A[8 + i] = ahi[i]; }
        v16bf B = *(const v16bf*)(wlow + (size_t)idx * LOWR + base + kbB);
        accL = __builtin_amdgcn_wmma_f32_16x16x32_bf16(false, A, false, B,
                                                       (short)0, accL, false, false);
    }
    int nout  = ntile * 16 + (lane & 15);
    int mbase = mtile * 16 + (lane >> 4) * 8;
    for (int rr = 0; rr < 8; ++rr) {
        int gmr = c * CHUNK + mbase + rr;
        float vm = acc[rr], va = accL[rr];
        if (idx == tids[gmr]) { vm = -__builtin_inff(); va = -__builtin_inff(); }
        negm[(size_t)gmr * NCAND + nout] = vm;
        nega[(size_t)gmr * NCAND + nout] = va;
    }
}

// ---- per-row online log-sum-exp loss
__device__ float row_lse_loss(const float* __restrict__ neg, float pos, int lane) {
    float p    = pos * SCALE;
    float rmax = (lane == 0) ? p : -__builtin_inff();   // seed with positive logit once
    float rsum = (lane == 0) ? 1.0f : 0.0f;
    for (int j = lane; j < NCAND; j += 32) {
        float x = neg[j] * SCALE;
        if (x < -1.0e30f) continue;                     // masked (-inf) entries
        if (x > rmax) { rsum = rsum * __expf(rmax - x) + 1.0f; rmax = x; }
        else          { rsum += __expf(x - rmax); }
    }
    for (int off = 16; off > 0; off >>= 1) {
        float om = __shfl_xor(rmax, off, 32);
        float os = __shfl_xor(rsum, off, 32);
        float nm = fmaxf(rmax, om);
        rsum = rsum * __expf(rmax - nm) + os * __expf(om - nm);
        rmax = nm;
    }
    return (rmax + __logf(rsum)) - p;                   // -log_softmax[:,0]
}

__global__ void loss_kernel(const float* __restrict__ hf,
                            const float* __restrict__ W,
                            const float* __restrict__ inv_full,
                            const __bf16* __restrict__ wlow,
                            const int* __restrict__ tids,
                            const float* __restrict__ negm,
                            const float* __restrict__ nega,
                            float* __restrict__ loss_chunk) {
    int c = blockIdx.x;
    int w = threadIdx.x >> 5, lane = threadIdx.x & 31;
    __shared__ float lm[32], la[32];
    int gm = c * CHUNK + w;
    int t  = tids[gm];
    const float* hrow = hf + (size_t)gm * DIM;
    const float* wrow = W  + (size_t)t  * DIM;
    float pf = 0.f, pl = 0.f;
    for (int i = 0; i < 32; ++i) {
        float hx = hrow[i * 32 + lane];
        pf += hx * wrow[i * 32 + lane];
        if (i < 2) pl += hx * (float)wlow[(size_t)t * LOWR + i * 32 + lane];
    }
    pf = wave_sum(pf) * inv_full[t];
    pl = wave_sum(pl);
    float lossm = row_lse_loss(negm + (size_t)gm * NCAND, pf, lane);
    float lossa = row_lse_loss(nega + (size_t)gm * NCAND, pl, lane);
    if (lane == 0) { lm[w] = lossm; la[w] = lossa; }
    __syncthreads();
    if (threadIdx.x == 0) {
        float sm = 0.f, sa = 0.f;
        for (int i = 0; i < CHUNK; ++i) { sm += lm[i]; sa += la[i]; }
        loss_chunk[c] = sm + AUXW * sa;
    }
}

__global__ void final_kernel(const float* __restrict__ loss_chunk, float* __restrict__ out) {
    if (threadIdx.x == 0 && blockIdx.x == 0) {
        float s = 0.f;
        for (int i = 0; i < NCHUNK; ++i) s += loss_chunk[i];   // fixed order -> deterministic
        out[0] = s / (float)NROWS;
    }
}

// ----------------------------------------------------------------
extern "C" void kernel_launch(void* const* d_in, const int* in_sizes, int n_in,
                              void* d_out, int out_size, void* d_ws, size_t ws_size,
                              hipStream_t stream) {
    const float* H = (const float*)d_in[0];   // (4096,1024)
    const float* W = (const float*)d_in[1];   // (50257,1024)
    const int*   T = (const int*)d_in[2];     // (4096,)
    (void)in_sizes; (void)n_in; (void)out_size;

    char* p = (char*)d_ws;
    auto alloc = [&](size_t bytes) -> char* {
        char* r = p;
        p += (bytes + 255) & ~(size_t)255;
        return r;
    };
    float*  hf     = (float*)alloc(sizeof(float) * (size_t)NROWS * DIM);
    __bf16* hb     = (__bf16*)alloc(sizeof(__bf16) * (size_t)NROWS * DIM);
    float*  hmean  = (float*)alloc(sizeof(float) * NCHUNK * LOWR);
    float*  invf   = (float*)alloc(sizeof(float) * VOCAB);
    float*  invl   = (float*)alloc(sizeof(float) * VOCAB);
    __bf16* wlow   = (__bf16*)alloc(sizeof(__bf16) * (size_t)VPAD * LOWR);
    __bf16* wnorm  = (__bf16*)alloc(sizeof(__bf16) * (size_t)VOCAB * DIM);
    float*  scan   = (float*)alloc(sizeof(float) * (size_t)NCHUNK * VPAD);
    int*    topidx = (int*)alloc(sizeof(int) * NCHUNK * NCAND);
    float*  negm   = (float*)alloc(sizeof(float) * (size_t)NROWS * NCAND);
    float*  nega   = (float*)alloc(sizeof(float) * (size_t)NROWS * NCAND);
    float*  lossc  = (float*)alloc(sizeof(float) * NCHUNK);
    if ((size_t)(p - (char*)d_ws) > ws_size) return;   // insufficient scratch

    emb_norm_kernel<<<(VPAD * 32 + 255) / 256, 256, 0, stream>>>(W, invf, invl, wlow, wnorm);
    h_norm_kernel<<<(NROWS * 32 + 255) / 256, 256, 0, stream>>>(H, hf, hb);
    h_mean_kernel<<<NCHUNK, LOWR, 0, stream>>>(hf, hmean);
    scan_kernel<<<VTILES, 32, 0, stream>>>(wlow, hmean, scan);
    topk_kernel<<<NCHUNK, 1024, 0, stream>>>(scan, topidx);
    neg_gemm_kernel<<<NCHUNK * 256, 32, 0, stream>>>(hb, wnorm, wlow, topidx, T, negm, nega);
    loss_kernel<<<NCHUNK, 1024, 0, stream>>>(hf, W, invf, wlow, T, negm, nega, lossc);
    final_kernel<<<1, 32, 0, stream>>>(lossc, (float*)d_out);
}
